// EfficientMultiHeadAttention_13451837571141
// MI455X (gfx1250) — matrix operand order repaired and checked by reading, not compiled, not run on hardware
//
#include <hip/hip_runtime.h>

typedef __attribute__((ext_vector_type(16))) _Float16 v16h;
typedef __attribute__((ext_vector_type(8)))  _Float16 v8h;
typedef __attribute__((ext_vector_type(8)))  float    v8f;

#define EMBED 64
#define SR    8
#define BATCH 16
#define HIMG  128
#define WIMG  128
#define NSEQ  (HIMG*WIMG)   /* 16384 */
#define NKV   256
#define EPSV  1e-5f
#define WAVES 4

// ---------------------------------------------------------------------------
// Kernel 1: spatial reduction + folded projections (tiny: 4096 blocks x 64 thr)
//   For each (batch, patch): conv8x8s8 -> LN -> k,v rows, then fold:
//     KqT[key][c] = sum_e k[e] * Wq[c][e]      (f16, B-operand layout rows)
//     VoT[o][key] = sum_c v[c] * Wo[c][o]      (f16)
//     kbias[key]  = sum_e bq[e] * k[e]
// ---------------------------------------------------------------------------
__global__ __launch_bounds__(64) void emha_prep(
    const float* __restrict__ X,   const float* __restrict__ Wq,
    const float* __restrict__ bq,  const float* __restrict__ Wkv,
    const float* __restrict__ bkv, const float* __restrict__ Wo,
    const float* __restrict__ convw, const float* __restrict__ convb,
    const float* __restrict__ gamma, const float* __restrict__ beta,
    _Float16* __restrict__ KqT, _Float16* __restrict__ VoT,
    float* __restrict__ kbias)
{
    __shared__ float xs[SR*SR*EMBED];                 // 16 KB patch
    __shared__ float cvals[EMBED], lnv[EMBED], kvals[EMBED], vvals[EMBED], red[EMBED];

    const int blk = blockIdx.x;
    const int b  = blk >> 8;
    const int p  = blk & 255;
    const int py = p >> 4, px = p & 15;
    const int o  = threadIdx.x;

    // stage the 8x8x64 input patch (coalesced: 64 lanes sweep channels)
    for (int s = 0; s < SR*SR; ++s) {
        int ky = s >> 3, kx = s & 7;
        size_t g = ((size_t)b*NSEQ + (size_t)(py*SR+ky)*WIMG + (px*SR+kx))*EMBED + o;
        xs[s*EMBED + o] = X[g];
    }
    __syncthreads();

    // conv output channel o (patch GEMM, K=4096)
    float acc = convb[o];
    for (int idx = 0; idx < SR*SR*EMBED; ++idx)
        acc = fmaf(xs[idx], convw[(size_t)idx*EMBED + o], acc);
    cvals[o] = acc;
    __syncthreads();

    // LayerNorm over 64 channels
    float mean = 0.f, var = 0.f;
    for (int i = 0; i < EMBED; ++i) mean += cvals[i];
    mean *= (1.f/EMBED);
    for (int i = 0; i < EMBED; ++i) { float d = cvals[i]-mean; var += d*d; }
    var *= (1.f/EMBED);
    float ln = (acc - mean) * rsqrtf(var + EPSV) * gamma[o] + beta[o];
    lnv[o] = ln;
    __syncthreads();

    // k, v rows
    float kk = bkv[o], vv = bkv[EMBED+o];
    for (int c = 0; c < EMBED; ++c) {
        float l = lnv[c];
        kk = fmaf(l, Wkv[c*2*EMBED + o],         kk);
        vv = fmaf(l, Wkv[c*2*EMBED + EMBED + o], vv);
    }
    kvals[o] = kk; vvals[o] = vv; red[o] = bq[o]*kk;
    __syncthreads();

    if (o == 0) {
        float s = 0.f;
        for (int i = 0; i < EMBED; ++i) s += red[i];
        kbias[b*NKV + p] = s;
    }

    // folded query projection:  KqT[key][c] = k . Wq[c,:]
    float sq = 0.f;
    for (int e = 0; e < EMBED; ++e) sq = fmaf(kvals[e], Wq[o*EMBED + e], sq);
    KqT[((size_t)b*NKV + p)*EMBED + o] = (_Float16)sq;

    // folded output projection: VoT[o][key] = v . Wo[:,o]
    float sv = 0.f;
    for (int c = 0; c < EMBED; ++c) sv = fmaf(vvals[c], Wo[c*EMBED + o], sv);
    VoT[((size_t)b*EMBED + o)*NKV + p] = (_Float16)sv;
}

// ---------------------------------------------------------------------------
// Kernel 2: fused attention. One wave per 16-query tile.
//   S = Xtile @ Kq   (32x v_wmma_f32_16x16x32_f16)
//   softmax in registers (shfl-xor row reductions, wave32 halves)
//   P -> LDS -> A-operand layout
//   O = P @ Vo       (32x v_wmma_f32_16x16x32_f16), + bo, store f32
// ---------------------------------------------------------------------------
__global__ __launch_bounds__(32*WAVES) void emha_attn(
    const float* __restrict__ X,
    const _Float16* __restrict__ KqT, const _Float16* __restrict__ VoT,
    const float* __restrict__ kbias, const float* __restrict__ bo,
    float* __restrict__ Out)
{
    __shared__ _Float16 Plds[WAVES*16*NKV];           // 32 KB
    const int lane = threadIdx.x & 31;
    const int wave = threadIdx.x >> 5;
    const int blocksPerBatch = (NSEQ/16)/WAVES;       // 256
    const int b    = blockIdx.x / blocksPerBatch;
    const int tile = (blockIdx.x % blocksPerBatch)*WAVES + wave;
    const int m0   = tile*16;

    const int nl    = lane & 15;       // A: row index ; B: column index
    const int half  = lane >> 4;       // wave32 half
    const int kbaseA = half*16;        // A layout: K base 0 / 16
    const int koffB  = half*16;        // B layout: K base 0 / 16

    // ---- A operand: X tile rows, f32->f16 on load -------------------------
    const float* xrow = X + ((size_t)b*NSEQ + (m0 + nl))*EMBED;
    v16h aX[2];
#pragma unroll
    for (int kc = 0; kc < 2; ++kc) {
        int k0 = kc*32 + kbaseA;
#pragma unroll
        for (int i = 0; i < 8; ++i) {
            aX[kc][i]   = (_Float16)xrow[k0 + i];
            aX[kc][i+8] = (_Float16)xrow[k0 + 8 + i];
        }
    }

    // ---- S = X @ Kq : 16 N-tiles x 2 K-steps ------------------------------
    v8f acc[16];
    const _Float16* kq = KqT + (size_t)b*NKV*EMBED;
#pragma unroll
    for (int nt = 0; nt < 16; ++nt) {
        v8f c = {};
        const _Float16* brow = kq + (size_t)(nt*16 + nl)*EMBED + koffB;
#pragma unroll
        for (int kc = 0; kc < 2; ++kc) {
            v16h bb = *reinterpret_cast<const v16h*>(brow + kc*32);
            c = __builtin_amdgcn_wmma_f32_16x16x32_f16(false, aX[kc], false, bb,
                                                       (short)0, c, false, false);
        }
        acc[nt] = c;
    }

    // ---- softmax over 256 keys, rows live in registers --------------------
    const float scale = 0.125f;        // 1/sqrt(64)
    float kb[16];
#pragma unroll
    for (int nt = 0; nt < 16; ++nt) kb[nt] = kbias[b*NKV + nt*16 + nl];

    _Float16* pw = Plds + wave*16*NKV;
#pragma unroll
    for (int r = 0; r < 8; ++r) {
        const int m = r + half*8;      // query row this (vgpr,half) holds
        float mx = -3.4e38f;
#pragma unroll
        for (int nt = 0; nt < 16; ++nt) {
            float l = (acc[nt][r] + kb[nt]) * scale;
            acc[nt][r] = l;
            mx = fmaxf(mx, l);
        }
#pragma unroll
        for (int off = 1; off < 16; off <<= 1)
            mx = fmaxf(mx, __shfl_xor(mx, off, 32));   // stays inside 16-lane half
        float sm = 0.f;
#pragma unroll
        for (int nt = 0; nt < 16; ++nt) {
            float e = __expf(acc[nt][r] - mx);
            acc[nt][r] = e;
            sm += e;
        }
#pragma unroll
        for (int off = 1; off < 16; off <<= 1)
            sm += __shfl_xor(sm, off, 32);
        float inv = 1.f / sm;
#pragma unroll
        for (int nt = 0; nt < 16; ++nt)
            pw[m*NKV + nt*16 + nl] = (_Float16)(acc[nt][r]*inv);
    }
    __syncthreads();

    // ---- reload P in A-operand layout -------------------------------------
    v16h aP[8];
#pragma unroll
    for (int kc = 0; kc < 8; ++kc) {
        int k0 = kc*32 + kbaseA;
        v8h lo = *reinterpret_cast<const v8h*>(pw + nl*NKV + k0);
        v8h hi = *reinterpret_cast<const v8h*>(pw + nl*NKV + k0 + 8);
#pragma unroll
        for (int i = 0; i < 8; ++i) { aP[kc][i] = lo[i]; aP[kc][i+8] = hi[i]; }
    }

    // ---- O = P @ Vo : 4 N-tiles x 8 K-steps -------------------------------
    const _Float16* vo = VoT + (size_t)b*EMBED*NKV;
    v8f oacc[4];
#pragma unroll
    for (int nt = 0; nt < 4; ++nt) {
        v8f c = {};
        const _Float16* brow = vo + (size_t)(nt*16 + nl)*NKV + koffB;
#pragma unroll
        for (int kc = 0; kc < 8; ++kc) {
            v16h bb = *reinterpret_cast<const v16h*>(brow + kc*32);
            c = __builtin_amdgcn_wmma_f32_16x16x32_f16(false, aP[kc], false, bb,
                                                       (short)0, c, false, false);
        }
        oacc[nt] = c;
    }

    // ---- + bo, store f32 (16 consecutive lanes -> 64B contiguous) ---------
#pragma unroll
    for (int nt = 0; nt < 4; ++nt) {
        float bb = bo[nt*16 + nl];
#pragma unroll
        for (int r = 0; r < 8; ++r) {
            int m = r + half*8;
            Out[((size_t)b*NSEQ + m0 + m)*EMBED + nt*16 + nl] = oacc[nt][r] + bb;
        }
    }
}

// ---------------------------------------------------------------------------
extern "C" void kernel_launch(void* const* d_in, const int* in_sizes, int n_in,
                              void* d_out, int out_size, void* d_ws, size_t ws_size,
                              hipStream_t stream)
{
    const float* X     = (const float*)d_in[0];
    const float* Wq    = (const float*)d_in[1];
    const float* bq    = (const float*)d_in[2];
    const float* Wkv   = (const float*)d_in[3];
    const float* bkv   = (const float*)d_in[4];
    const float* Wo    = (const float*)d_in[5];
    const float* bo    = (const float*)d_in[6];
    const float* convw = (const float*)d_in[7];
    const float* convb = (const float*)d_in[8];
    const float* gamma = (const float*)d_in[9];
    const float* beta  = (const float*)d_in[10];
    float* Out = (float*)d_out;

    // workspace: KqT f16 (512KB) | VoT f16 (512KB) | kbias f32 (16KB)
    char* ws = (char*)d_ws;
    _Float16* KqT  = (_Float16*)(ws);
    _Float16* VoT  = (_Float16*)(ws + (size_t)BATCH*NKV*EMBED*sizeof(_Float16));
    float*    kbia = (float*)   (ws + (size_t)2*BATCH*NKV*EMBED*sizeof(_Float16));

    emha_prep<<<BATCH*NKV, 64, 0, stream>>>(X, Wq, bq, Wkv, bkv, Wo, convw, convb,
                                            gamma, beta, KqT, VoT, kbia);

    const int grid = BATCH * ((NSEQ/16)/WAVES);      // 4096 blocks
    emha_attn<<<grid, 32*WAVES, 0, stream>>>(X, KqT, VoT, kbia, bo, Out);
}
